// GatedSpatioTempAttention_31456340476274
// MI455X (gfx1250) — matrix operand order repaired and checked by reading, not compiled, not run on hardware
//
#include <hip/hip_runtime.h>
#include <hip/hip_bf16.h>
#include <math.h>
#include <stdint.h>

// GatedSpatioTempAttention for MI455X (gfx1250), bf16 WMMA pipeline.
// B=8, N=2048, H=1024, 2H=2048.

typedef __attribute__((ext_vector_type(16))) __bf16 v16bf;
typedef __attribute__((ext_vector_type(8)))  __bf16 v8bf;
typedef __attribute__((ext_vector_type(4)))  __bf16 v4bf;
typedef __attribute__((ext_vector_type(8)))  float  v8f;
typedef __attribute__((ext_vector_type(4)))  float  v4f;

#define HDIM   1024
#define H2DIM  2048
#define NSEQ   2048
#define NBAT   8
#define NROWS  (NBAT * NSEQ)   // 16384

// ---- WMMA fragment helpers (wave32, 16x16x32 bf16) -------------------------
// A (16x32 bf16): lane l(0..15)/half: row = l, elems 0..7  = K[k0 + half*8 .. +7]
//                                            elems 8..15 = K[k0+16+half*8 .. +7]
__device__ __forceinline__ v16bf load_a_frag(const __bf16* p0, const __bf16* p1) {
  v8bf lo = *(const v8bf*)p0;
  v8bf hi = *(const v8bf*)p1;
  return __builtin_shufflevector(lo, hi, 0,1,2,3,4,5,6,7,8,9,10,11,12,13,14,15);
}
// B (32x16 bf16): lane col = l, 16 contiguous K at k0 + half*16 (32B aligned)
__device__ __forceinline__ v16bf load_b_frag(const __bf16* p) {
  return *(const v16bf*)p;
}
__device__ __forceinline__ v8f wmma_bf16(v16bf a, v16bf b, v8f c) {
  return __builtin_amdgcn_wmma_f32_16x16x32_bf16(false, a, false, b, (short)0, c,
                                                 false, false);
}

// ---- 1) pack sem||temp into bf16 combined [16384 x 2048] -------------------
__global__ void k_convert(const float* __restrict__ sem,
                          const float* __restrict__ temp,
                          __bf16* __restrict__ comb) {
  size_t i = ((size_t)blockIdx.x * 256 + threadIdx.x) * 8; // over 16384*1024
  size_t row = i >> 10;          // /H
  size_t c   = i & 1023;         // %H
  v4f s0 = *(const v4f*)(sem + i);
  v4f s1 = *(const v4f*)(sem + i + 4);
  v4f t0 = *(const v4f*)(temp + i);
  v4f t1 = *(const v4f*)(temp + i + 4);
  v8bf sb, tb;
#pragma unroll
  for (int j = 0; j < 4; ++j) {
    sb[j] = (__bf16)s0[j]; sb[4 + j] = (__bf16)s1[j];
    tb[j] = (__bf16)t0[j]; tb[4 + j] = (__bf16)t1[j];
  }
  *(v8bf*)(comb + row * H2DIM + c)        = sb;
  *(v8bf*)(comb + row * H2DIM + HDIM + c) = tb;
}

// ---- 2) W [2H x H] f32 -> Wt [H x 2H] bf16 (B-fragment friendly) -----------
__global__ void k_tw(const float* __restrict__ W, __bf16* __restrict__ Wt) {
  int idx = blockIdx.x * 256 + threadIdx.x;   // over 2048*1024
  int k = idx >> 10;
  int n = idx & 1023;
  Wt[(size_t)n * H2DIM + k] = (__bf16)W[idx];
}

// ---- 3) QKV GEMM: [16384 x 2048] x [2048 x 1024] (+bias) -------------------
// mode 0/1: row-major bf16 out [16384 x 1024]; mode 2: Vt[b][h][n] bf16.
__global__ __launch_bounds__(128, 1)
void k_qkv(const __bf16* __restrict__ A, const __bf16* __restrict__ Wt,
           const float* __restrict__ bias, __bf16* __restrict__ out, int mode) {
  const int tid  = threadIdx.x;
  const int wave = tid >> 5;
  const int lane = tid & 31;
  const int l    = lane & 15;
  const int half = lane >> 4;
  const int m0 = blockIdx.x * 128 + (wave >> 1) * 64;
  const int n0 = blockIdx.y * 128 + (wave & 1) * 64;

  const __bf16* arow[4];
#pragma unroll
  for (int mi = 0; mi < 4; ++mi)
    arow[mi] = A + (size_t)(m0 + mi * 16 + l) * H2DIM;
  const __bf16* bcol = Wt + (size_t)(n0 + l) * H2DIM;

  v8f acc[4][4];
#pragma unroll
  for (int mi = 0; mi < 4; ++mi)
#pragma unroll
    for (int ni = 0; ni < 4; ++ni)
#pragma unroll
      for (int r = 0; r < 8; ++r) acc[mi][ni][r] = 0.f;

  for (int kk = 0; kk < H2DIM; kk += 32) {
    v16bf a[4];
#pragma unroll
    for (int mi = 0; mi < 4; ++mi)
      a[mi] = load_a_frag(arow[mi] + kk + half * 8,
                          arow[mi] + kk + 16 + half * 8);
#pragma unroll
    for (int ni = 0; ni < 4; ++ni) {
      v16bf bf = load_a_frag(bcol + (size_t)ni * 16 * H2DIM + kk + half * 8,
                             bcol + (size_t)ni * 16 * H2DIM + kk + 16 + half * 8);
#pragma unroll
      for (int mi = 0; mi < 4; ++mi)
        acc[mi][ni] = wmma_bf16(a[mi], bf, acc[mi][ni]);
    }
  }

  if (mode == 2) {
    // V transposed: Vt[b][h][q]; rows r=0..7 are 8 consecutive q -> one v8bf store
#pragma unroll
    for (int ni = 0; ni < 4; ++ni) {
      int n = n0 + ni * 16 + l;
      float bv = bias[n];
#pragma unroll
      for (int mi = 0; mi < 4; ++mi) {
        int mbase  = m0 + mi * 16;
        int bb     = mbase >> 11;
        int qstart = (mbase & 2047) + 8 * half;
        v8bf vv;
#pragma unroll
        for (int r = 0; r < 8; ++r) vv[r] = (__bf16)(acc[mi][ni][r] + bv);
        *(v8bf*)(out + ((size_t)bb << 21) + (size_t)n * NSEQ + qstart) = vv;
      }
    }
  } else {
#pragma unroll
    for (int ni = 0; ni < 4; ++ni) {
      int n = n0 + ni * 16 + l;
      float bv = bias[n];
#pragma unroll
      for (int mi = 0; mi < 4; ++mi) {
#pragma unroll
        for (int r = 0; r < 8; ++r) {
          int m = m0 + mi * 16 + r + 8 * half;
          out[(size_t)m * HDIM + n] = (__bf16)(acc[mi][ni][r] + bv);
        }
      }
    }
  }
}

// ---- 4) gate[row] = sigmoid(combined . Wg + bg) (f32, tiny) ----------------
__global__ void k_gate(const float* __restrict__ sem, const float* __restrict__ temp,
                       const float* __restrict__ Wg, const float* __restrict__ bg,
                       float* __restrict__ gate) {
  __shared__ float red[256];
  int row = blockIdx.x, t = threadIdx.x;
  const float* srow = sem + (size_t)row * HDIM;
  const float* trow = temp + (size_t)row * HDIM;
  float s = 0.f;
  for (int j = t; j < HDIM; j += 256)
    s += srow[j] * Wg[j] + trow[j] * Wg[HDIM + j];
  red[t] = s;
  __syncthreads();
  for (int off = 128; off > 0; off >>= 1) {
    if (t < off) red[t] += red[t + off];
    __syncthreads();
  }
  if (t == 0) gate[row] = 1.f / (1.f + __expf(-(red[0] + bg[0])));
}

__global__ void k_zero(float* __restrict__ p, int n) {
  int i = blockIdx.x * 256 + threadIdx.x;
  if (i < n) p[i] = 0.f;
}

// ---- 5) attention: scores -> softmax*gate -> P.V (+sem), colsum ------------
// grid (N/16, B), 256 threads = 8 waves. Wave w owns key-cols [w*256, w*256+256)
// for GEMM1 and h-cols [w*128, w*128+128) for GEMM2.
__global__ __launch_bounds__(256, 1)
void k_attn(const __bf16* __restrict__ Q, const __bf16* __restrict__ Kb,
            const __bf16* __restrict__ Vt, const float* __restrict__ gate,
            const float* __restrict__ sem, float* __restrict__ outF,
            float* __restrict__ colsum) {
  constexpr int QS = 1032;  // padded Q-tile stride (bank-conflict-free ds_load_b128)
  constexpr int PS = 2056;  // padded probs stride
  __shared__ __attribute__((aligned(16))) __bf16 qtile[16 * QS];
  __shared__ __attribute__((aligned(16))) __bf16 probs[16 * PS];
  __shared__ float red[8 * 16];

  const int tid  = threadIdx.x;
  const int wave = tid >> 5;
  const int lane = tid & 31;
  const int l    = lane & 15;
  const int half = lane >> 4;
  const int b  = blockIdx.y;
  const int q0 = blockIdx.x * 16;

  // ---- async-stage Q tile (16 x 1024 bf16 = 32 KB) into LDS ----
  {
    const __bf16* qg = Q + (size_t)(b * NSEQ + q0) * HDIM;
    for (int c = tid; c < 2048; c += 256) {   // 2048 x 16B chunks
      int row = c >> 7;                        // 128 chunks per row
      int col = (c & 127) * 8;                 // bf16 elems
      unsigned lds  = (unsigned)(uintptr_t)&qtile[row * QS + col];
      const __bf16* g = qg + (size_t)row * HDIM + col;
      asm volatile("global_load_async_to_lds_b128 %0, %1, off"
                   :: "v"(lds), "v"(g) : "memory");
    }
    asm volatile("s_wait_asynccnt 0" ::: "memory");
  }
  __syncthreads();

  const __bf16* kbase = Kb + (size_t)(b * NSEQ + wave * 256 + l) * HDIM;

  // ---- GEMM1: scores[16 x 256] = Q_tile x K^T ----
  v8f acc[16];
#pragma unroll
  for (int t = 0; t < 16; ++t)
#pragma unroll
    for (int r = 0; r < 8; ++r) acc[t][r] = 0.f;

  for (int kk = 0; kk < HDIM; kk += 32) {
    v16bf a = load_a_frag(&qtile[l * QS + kk + half * 8],
                          &qtile[l * QS + kk + 16 + half * 8]);
#pragma unroll
    for (int t = 0; t < 16; ++t) {
      v16bf bf = load_a_frag(kbase + (size_t)t * 16 * HDIM + kk + half * 8,
                             kbase + (size_t)t * 16 * HDIM + kk + 16 + half * 8);
      acc[t] = wmma_bf16(a, bf, acc[t]);
    }
  }
  const float scale = 0.03125f;  // 1/sqrt(1024)
#pragma unroll
  for (int t = 0; t < 16; ++t)
#pragma unroll
    for (int r = 0; r < 8; ++r) acc[t][r] *= scale;

  // ---- softmax across 2048 keys (cross-wave via LDS) ----
  // Lane-half owns rows r+8*half; per-lane values span 16 tiles.
  float gmax[8], gsum[8];
#pragma unroll
  for (int r = 0; r < 8; ++r) {
    float m = acc[0][r];
#pragma unroll
    for (int t = 1; t < 16; ++t) m = fmaxf(m, acc[t][r]);
#pragma unroll
    for (int s = 1; s < 16; s <<= 1) m = fmaxf(m, __shfl_xor(m, s, 32));
    if (l == 0) red[wave * 16 + r + 8 * half] = m;
  }
  __syncthreads();
#pragma unroll
  for (int r = 0; r < 8; ++r) {
    float m = red[r + 8 * half];
#pragma unroll
    for (int wv = 1; wv < 8; ++wv) m = fmaxf(m, red[wv * 16 + r + 8 * half]);
    gmax[r] = m;
  }
  __syncthreads();
#pragma unroll
  for (int r = 0; r < 8; ++r) {
    float s = 0.f;
#pragma unroll
    for (int t = 0; t < 16; ++t) {
      float e = __expf(acc[t][r] - gmax[r]);
      acc[t][r] = e;
      s += e;
    }
#pragma unroll
    for (int sh = 1; sh < 16; sh <<= 1) s += __shfl_xor(s, sh, 32);
    if (l == 0) red[wave * 16 + r + 8 * half] = s;
  }
  __syncthreads();
#pragma unroll
  for (int r = 0; r < 8; ++r) {
    float s = 0.f;
#pragma unroll
    for (int wv = 0; wv < 8; ++wv) s += red[wv * 16 + r + 8 * half];
    gsum[r] = 1.f / s;
  }

  // ---- gated probs -> LDS (bf16) + column-mean accumulation ----
  float g[8];
#pragma unroll
  for (int r = 0; r < 8; ++r) g[r] = gate[b * NSEQ + q0 + r + 8 * half];

#pragma unroll
  for (int t = 0; t < 16; ++t) {
    int n = wave * 256 + t * 16 + l;
    float cs = 0.f;
#pragma unroll
    for (int r = 0; r < 8; ++r) {
      float p = acc[t][r] * gsum[r] * g[r];
      cs += p;
      probs[(r + 8 * half) * PS + n] = (__bf16)p;
    }
    cs += __shfl_xor(cs, 16, 32);  // combine rows 0-7 and 8-15 halves
    if (half == 0)
      atomicAdd(&colsum[b * NSEQ + n], cs * (1.0f / 2048.0f));
  }
  __syncthreads();

  // ---- GEMM2: fused[16 x 128] = P x V (B-frags from Vt rows) ----
  v8f acc2[8];
#pragma unroll
  for (int ni = 0; ni < 8; ++ni)
#pragma unroll
    for (int r = 0; r < 8; ++r) acc2[ni][r] = 0.f;

  const __bf16* vbase = Vt + ((size_t)b << 21) + (size_t)(wave * 128 + l) * NSEQ;
  for (int kk = 0; kk < NSEQ; kk += 32) {
    v16bf a = load_a_frag(&probs[l * PS + kk + half * 8],
                          &probs[l * PS + kk + 16 + half * 8]);
#pragma unroll
    for (int ni = 0; ni < 8; ++ni) {
      v16bf bf = load_a_frag(vbase + (size_t)ni * 16 * NSEQ + kk + half * 8,
                             vbase + (size_t)ni * 16 * NSEQ + kk + 16 + half * 8);
      acc2[ni] = wmma_bf16(a, bf, acc2[ni]);
    }
  }

  // ---- epilogue: out = fused + sem ----
#pragma unroll
  for (int ni = 0; ni < 8; ++ni) {
    int hcol = wave * 128 + ni * 16 + l;
#pragma unroll
    for (int r = 0; r < 8; ++r) {
      int q = q0 + r + 8 * half;
      size_t idx = (size_t)(b * NSEQ + q) * HDIM + hcol;
      outF[idx] = acc2[ni][r] + sem[idx];
    }
  }
}

// ---- host-side orchestration ----------------------------------------------
extern "C" void kernel_launch(void* const* d_in, const int* in_sizes, int n_in,
                              void* d_out, int out_size, void* d_ws, size_t ws_size,
                              hipStream_t stream) {
  (void)in_sizes; (void)n_in; (void)out_size; (void)ws_size;
  const float* sem  = (const float*)d_in[0];
  const float* temp = (const float*)d_in[1];
  const float* Wq   = (const float*)d_in[2];
  const float* bq   = (const float*)d_in[3];
  const float* Wk   = (const float*)d_in[4];
  const float* bk   = (const float*)d_in[5];
  const float* Wv   = (const float*)d_in[6];
  const float* bv   = (const float*)d_in[7];
  const float* Wg   = (const float*)d_in[8];
  const float* bg   = (const float*)d_in[9];

  char* ws = (char*)d_ws;
  __bf16* comb = (__bf16*)ws;  ws += (size_t)NROWS * H2DIM * 2;  // 67.1 MB
  __bf16* WqT  = (__bf16*)ws;  ws += (size_t)HDIM * H2DIM * 2;   // 4.2 MB
  __bf16* WkT  = (__bf16*)ws;  ws += (size_t)HDIM * H2DIM * 2;
  __bf16* WvT  = (__bf16*)ws;  ws += (size_t)HDIM * H2DIM * 2;
  __bf16* Qb   = (__bf16*)ws;  ws += (size_t)NROWS * HDIM * 2;   // 33.6 MB
  __bf16* Kbm  = (__bf16*)ws;  ws += (size_t)NROWS * HDIM * 2;
  __bf16* Vt   = (__bf16*)ws;  ws += (size_t)NROWS * HDIM * 2;
  float*  gateb = (float*)ws;  ws += (size_t)NROWS * 4;

  float* outF   = (float*)d_out;
  float* colsum = outF + (size_t)NROWS * HDIM;   // [B, N] means

  k_convert<<<8192, 256, 0, stream>>>(sem, temp, comb);
  k_tw<<<8192, 256, 0, stream>>>(Wq, WqT);
  k_tw<<<8192, 256, 0, stream>>>(Wk, WkT);
  k_tw<<<8192, 256, 0, stream>>>(Wv, WvT);

  dim3 gq(128, 8);
  k_qkv<<<gq, 128, 0, stream>>>(comb, WqT, bq, Qb, 0);
  k_qkv<<<gq, 128, 0, stream>>>(comb, WkT, bk, Kbm, 1);
  k_qkv<<<gq, 128, 0, stream>>>(comb, WvT, bv, Vt, 2);

  k_gate<<<16384, 256, 0, stream>>>(sem, temp, Wg, bg, gateb);
  k_zero<<<64, 256, 0, stream>>>(colsum, NROWS);

  dim3 ga(128, 8);
  k_attn<<<ga, 256, 0, stream>>>(Qb, Kbm, Vt, gateb, sem, outF, colsum);
}
